// Encoder_24610162606527
// MI455X (gfx1250) — matrix operand order repaired and checked by reading, not compiled, not run on hardware
//
#include <hip/hip_runtime.h>

#define IN_CH  512
#define HID    256
#define OUT_CH 128

typedef __attribute__((ext_vector_type(16))) __bf16 v16bf;
typedef __attribute__((ext_vector_type(8)))  float  v8f;

union FragBF { v16bf v; unsigned u[8]; };
union BFBits { __bf16 h; unsigned short s; };

__device__ inline unsigned short bf_hi(float v) { BFBits b; b.h = (__bf16)v; return b.s; }
__device__ inline float bf_hi_f(float v)        { BFBits b; b.h = (__bf16)v; return (float)b.h; }

// ---------------------------------------------------------------------------
// Tiled bf16-split WMMA GEMM:  out = A[MxK] @ B[KxN] (+bias), fp32 in/out.
// a = hi + lo (bf16 each); product = hi*hi + hi*lo + lo*hi (3 WMMAs).
// Column space [0,nsplit) -> (B0,out0,bias0); [nsplit,..) -> (B1,out1,bias1).
// BN=128 aligns with the split so selection is block-uniform.
// ---------------------------------------------------------------------------
__global__ __launch_bounds__(256)
void gemm_wmma_bf16split(const float* __restrict__ A, int M, int K, int lda,
                         const float* __restrict__ B0, const float* __restrict__ B1,
                         int ldb, int nsplit,
                         float* __restrict__ out0, float* __restrict__ out1, int ldo,
                         const float* __restrict__ bias0, const float* __restrict__ bias1)
{
    const int STR = 34;                       // padded LDS stride (halves), even
    __shared__ unsigned short sAhi[128 * STR];
    __shared__ unsigned short sAlo[128 * STR];
    __shared__ unsigned short sBhi[128 * STR];
    __shared__ unsigned short sBlo[128 * STR];

    const int tid     = threadIdx.x;
    const int mBase   = blockIdx.y * 128;
    const int colBase = blockIdx.x * 128;

    const float* Bp;  const float* biasp;  float* outp;  int bcoff;
    if (colBase < nsplit) { Bp = B0; biasp = bias0; outp = out0; bcoff = colBase; }
    else                  { Bp = B1; biasp = bias1; outp = out1; bcoff = colBase - nsplit; }

    const int wave  = tid >> 5, lane = tid & 31;
    const int waveM = wave & 3, waveN = wave >> 2;     // 4x2 wave grid
    const int lhalf = lane >> 4, l16 = lane & 15;

    v8f zero = {};
    v8f acc[2][4];
    #pragma unroll
    for (int i = 0; i < 2; ++i)
        #pragma unroll
        for (int j = 0; j < 4; ++j) acc[i][j] = zero;

    for (int k0 = 0; k0 < K; k0 += 32) {
        __syncthreads();
        // ---- stage A tile 128x32 as float4 (1024 vec4, 4 per thread), [m][k]
        #pragma unroll
        for (int it = 0; it < 4; ++it) {
            int idx = tid + it * 256;               // 0..1023
            int m  = idx >> 3;
            int k4 = (idx & 7) << 2;
            int gm = mBase + m;
            float4 v = make_float4(0.f, 0.f, 0.f, 0.f);
            if (gm < M) v = *(const float4*)&A[(size_t)gm * lda + (k0 + k4)];
            float h0 = bf_hi_f(v.x), h1 = bf_hi_f(v.y), h2 = bf_hi_f(v.z), h3 = bf_hi_f(v.w);
            unsigned hi01 = (unsigned)bf_hi(v.x) | ((unsigned)bf_hi(v.y) << 16);
            unsigned hi23 = (unsigned)bf_hi(v.z) | ((unsigned)bf_hi(v.w) << 16);
            unsigned lo01 = (unsigned)bf_hi(v.x - h0) | ((unsigned)bf_hi(v.y - h1) << 16);
            unsigned lo23 = (unsigned)bf_hi(v.z - h2) | ((unsigned)bf_hi(v.w - h3) << 16);
            *(unsigned*)&sAhi[m * STR + k4]     = hi01;
            *(unsigned*)&sAhi[m * STR + k4 + 2] = hi23;
            *(unsigned*)&sAlo[m * STR + k4]     = lo01;
            *(unsigned*)&sAlo[m * STR + k4 + 2] = lo23;
        }
        // ---- stage B tile 32x128 as float4 along n, transposed into [n][k]
        #pragma unroll
        for (int it = 0; it < 4; ++it) {
            int idx = tid + it * 256;               // 0..1023
            int k  = idx >> 5;
            int n4 = (idx & 31) << 2;
            float4 v = *(const float4*)&Bp[(size_t)(k0 + k) * ldb + (bcoff + n4)];
            float h0 = bf_hi_f(v.x), h1 = bf_hi_f(v.y), h2 = bf_hi_f(v.z), h3 = bf_hi_f(v.w);
            sBhi[(n4 + 0) * STR + k] = bf_hi(v.x);
            sBhi[(n4 + 1) * STR + k] = bf_hi(v.y);
            sBhi[(n4 + 2) * STR + k] = bf_hi(v.z);
            sBhi[(n4 + 3) * STR + k] = bf_hi(v.w);
            sBlo[(n4 + 0) * STR + k] = bf_hi(v.x - h0);
            sBlo[(n4 + 1) * STR + k] = bf_hi(v.y - h1);
            sBlo[(n4 + 2) * STR + k] = bf_hi(v.z - h2);
            sBlo[(n4 + 3) * STR + k] = bf_hi(v.w - h3);
        }
        __syncthreads();

        // ---- A fragments (16-bit A 16x32 layout: half 0 K{0..7,16..23}, half 1 +8)
        FragBF ahi[2], alo[2];
        #pragma unroll
        for (int ms = 0; ms < 2; ++ms) {
            int row = waveM * 32 + ms * 16 + l16;
            const unsigned* ph = (const unsigned*)&sAhi[row * STR + 8 * lhalf];
            const unsigned* pl = (const unsigned*)&sAlo[row * STR + 8 * lhalf];
            #pragma unroll
            for (int i = 0; i < 4; ++i) { ahi[ms].u[i]     = ph[i];     alo[ms].u[i]     = pl[i]; }
            #pragma unroll
            for (int i = 0; i < 4; ++i) { ahi[ms].u[4 + i] = ph[8 + i]; alo[ms].u[4 + i] = pl[8 + i]; }
        }
        // ---- B fragments: lane holds col N=l16, K {0..7,16..23} + 8*lhalf
        #pragma unroll
        for (int ns = 0; ns < 4; ++ns) {
            int nrow = waveN * 64 + ns * 16 + l16;
            FragBF bhi, blo;
            const unsigned* ph = (const unsigned*)&sBhi[nrow * STR + 8 * lhalf];
            const unsigned* pl = (const unsigned*)&sBlo[nrow * STR + 8 * lhalf];
            #pragma unroll
            for (int i = 0; i < 8; ++i) { bhi.u[i] = ph[i + (i >> 2) * 4]; blo.u[i] = pl[i + (i >> 2) * 4]; }
            #pragma unroll
            for (int ms = 0; ms < 2; ++ms) {
                acc[ms][ns] = __builtin_amdgcn_wmma_f32_16x16x32_bf16(
                    false, ahi[ms].v, false, bhi.v, (short)0, acc[ms][ns], false, false);
                acc[ms][ns] = __builtin_amdgcn_wmma_f32_16x16x32_bf16(
                    false, ahi[ms].v, false, blo.v, (short)0, acc[ms][ns], false, false);
                acc[ms][ns] = __builtin_amdgcn_wmma_f32_16x16x32_bf16(
                    false, alo[ms].v, false, bhi.v, (short)0, acc[ms][ns], false, false);
            }
        }
    }

    // ---- epilogue: C/D layout — reg r, lane: M = r + 8*lhalf, N = l16
    #pragma unroll
    for (int ms = 0; ms < 2; ++ms)
        #pragma unroll
        for (int ns = 0; ns < 4; ++ns)
            #pragma unroll
            for (int r = 0; r < 8; ++r) {
                int grow = mBase + waveM * 32 + ms * 16 + r + 8 * lhalf;
                if (grow < M) {
                    int ncol = bcoff + waveN * 64 + ns * 16 + l16;
                    float v = acc[ms][ns][r];
                    if (biasp) v += biasp[ncol];
                    outp[(size_t)grow * ldo + ncol] = v;
                }
            }
}

// ---------------------------------------------------------------------------
// Support kernels
// ---------------------------------------------------------------------------
__global__ void k_fill_f32(float* p, float v, int n) {
    int i = blockIdx.x * blockDim.x + threadIdx.x;
    if (i < n) p[i] = v;
}
__global__ void k_fill_u32(unsigned* p, unsigned v, int n) {
    int i = blockIdx.x * blockDim.x + threadIdx.x;
    if (i < n) p[i] = v;
}
__global__ void k_deg(const long long* __restrict__ dst, unsigned* __restrict__ deg, int E) {
    int e = blockIdx.x * blockDim.x + threadIdx.x;
    if (e < E) atomicAdd(&deg[(int)dst[e]], 1u);
}
__global__ void k_dis(const unsigned* __restrict__ deg, float* __restrict__ dis, int n) {
    int i = blockIdx.x * blockDim.x + threadIdx.x;
    if (i < n) dis[i] = rsqrtf((float)deg[i] + 1.0f);   // +1 self-loop
}
// one wave per edge; float4 gather (2x b128/lane) + 8 fp32 atomics/lane
__global__ __launch_bounds__(256)
void k_scatter(const long long* __restrict__ src, const long long* __restrict__ dst,
               const float* __restrict__ dis, const float* __restrict__ h,
               float* __restrict__ agg, int E)
{
    int e = blockIdx.x * 8 + (threadIdx.x >> 5);
    if (e >= E) return;
    int lane = threadIdx.x & 31;
    int s = (int)src[e], d = (int)dst[e];
    float nrm = dis[s] * dis[d];
    const float4* hp = (const float4*)(h + (size_t)s * HID);
    float* ap = agg + (size_t)d * HID;
    #pragma unroll
    for (int rep = 0; rep < 2; ++rep) {
        int v4 = lane + rep * 32;                 // float4 index 0..63
        float4 v = hp[v4];
        int c = v4 << 2;
        unsafeAtomicAdd(&ap[c + 0], v.x * nrm);
        unsafeAtomicAdd(&ap[c + 1], v.y * nrm);
        unsafeAtomicAdd(&ap[c + 2], v.z * nrm);
        unsafeAtomicAdd(&ap[c + 3], v.w * nrm);
    }
}
// agg[i][c] += dis[i]^2 * h[i][c] (+bias) (optional relu), in place; float4
__global__ void k_selfloop(float4* __restrict__ agg, const float4* __restrict__ h,
                           const float* __restrict__ dis, const float4* __restrict__ bias,
                           int Nn, int relu)
{
    int idx = blockIdx.x * blockDim.x + threadIdx.x;
    if (idx >= Nn * (HID / 4)) return;
    int i  = idx >> 6;            // HID/4 == 64
    int c4 = idx & 63;
    float d2 = dis[i] * dis[i];
    float4 a  = agg[idx];
    float4 hv = h[idx];
    a.x += d2 * hv.x; a.y += d2 * hv.y; a.z += d2 * hv.z; a.w += d2 * hv.w;
    if (bias) {
        float4 b = bias[c4];
        a.x += b.x; a.y += b.y; a.z += b.z; a.w += b.w;
    }
    if (relu) {
        a.x = fmaxf(a.x, 0.f); a.y = fmaxf(a.y, 0.f);
        a.z = fmaxf(a.z, 0.f); a.w = fmaxf(a.w, 0.f);
    }
    agg[idx] = a;
}

// ---------------------------------------------------------------------------
extern "C" void kernel_launch(void* const* d_in, const int* in_sizes, int n_in,
                              void* d_out, int out_size, void* d_ws, size_t ws_size,
                              hipStream_t stream)
{
    (void)n_in; (void)out_size; (void)ws_size;
    const float*     x   = (const float*)d_in[0];
    const long long* ei  = (const long long*)d_in[1];
    const float*     W1  = (const float*)d_in[2];
    const float*     b1  = (const float*)d_in[3];
    const float*     Wmu = (const float*)d_in[4];
    const float*     bmu = (const float*)d_in[5];
    const float*     Wls = (const float*)d_in[6];
    const float*     bls = (const float*)d_in[7];

    const int N = in_sizes[0] / IN_CH;
    const int E = in_sizes[1] / 2;
    const long long* src = ei;
    const long long* dst = ei + E;

    char* ws = (char*)d_ws;
    unsigned* deg = (unsigned*)ws;  ws += (size_t)N * 4;
    float* dis    = (float*)ws;     ws += (size_t)N * 4;
    float* bufA   = (float*)ws;     ws += (size_t)N * HID * 4;   // h1, then aggh
    float* bufB   = (float*)ws;                                   // agg1, then h

    float* mu = (float*)d_out;
    float* ls = mu + (size_t)N * OUT_CH;

    const int t   = 256;
    const int nc  = N * HID;
    const int nc4 = N * (HID / 4);

    // degrees + symmetric norm
    k_fill_u32<<<(N + t - 1) / t, t, 0, stream>>>(deg, 0u, N);
    k_deg<<<(E + t - 1) / t, t, 0, stream>>>(dst, deg, E);
    k_dis<<<(N + t - 1) / t, t, 0, stream>>>(deg, dis, N);

    dim3 gg(2, (unsigned)((N + 127) / 128));

    // h1 = x @ W1  (no bias yet; bias applied after aggregation)
    gemm_wmma_bf16split<<<gg, 256, 0, stream>>>(
        x, N, IN_CH, IN_CH, W1, nullptr, HID, HID,
        bufA, nullptr, HID, nullptr, nullptr);

    // h = relu(A_hat h1 + b1)
    k_fill_f32<<<(nc + t - 1) / t, t, 0, stream>>>(bufB, 0.0f, nc);
    k_scatter<<<(E + 7) / 8, 256, 0, stream>>>(src, dst, dis, bufA, bufB, E);
    k_selfloop<<<(nc4 + t - 1) / t, t, 0, stream>>>(
        (float4*)bufB, (const float4*)bufA, dis, (const float4*)b1, N, 1);

    // aggh = A_hat h   (shared between mu and logstd branches)
    k_fill_f32<<<(nc + t - 1) / t, t, 0, stream>>>(bufA, 0.0f, nc);
    k_scatter<<<(E + 7) / 8, 256, 0, stream>>>(src, dst, dis, bufB, bufA, E);
    k_selfloop<<<(nc4 + t - 1) / t, t, 0, stream>>>(
        (float4*)bufA, (const float4*)bufB, dis, nullptr, N, 0);

    // [mu | logstd] = aggh @ [Wmu | Wls] + [bmu | bls]
    gemm_wmma_bf16split<<<gg, 256, 0, stream>>>(
        bufA, N, HID, HID, Wmu, Wls, OUT_CH, OUT_CH,
        mu, ls, OUT_CH, bmu, bls);
}
